// DPEncoderBlock_17068200034328
// MI455X (gfx1250) — matrix operand order repaired and checked by reading, not compiled, not run on hardware
//
#include <hip/hip_runtime.h>

// ---------------------------------------------------------------------------
// Transformer encoder block for gfx1250 (MI455X), bf16 WMMA everywhere.
//   B=4, S=1024, D=1024, H=16, DH=64, DFF=4096, fp32 I/O.
// - weights transposed (N-major) once, all fragments = 2x ds_load_b128
// - global->LDS staging via gfx1250 async DMA (ASYNCcnt), double-buffered
// - GEMM: 128x128 block, 4 waves, 64x64 per wave (16 WMMA : 8 frag loads)
// - flash attention, 1/sqrt(dh) folded into Q projection epilogue
// - native bf16 converts (v_cvt, RNE)
// ---------------------------------------------------------------------------

#define TOK   4096      // B*S
#define DIM   1024
#define NDFF  4096
#define NH    16
#define DH    64
#define SEQ   1024
#define NB    4

typedef unsigned short u16;
typedef __bf16 bf16_t;
typedef bf16_t v16bf __attribute__((ext_vector_type(16)));
typedef float  v8f   __attribute__((ext_vector_type(8)));
typedef int    v4i   __attribute__((ext_vector_type(4)));

union BF16Frag { v16bf v; u16 u[16]; };

// ---- CDNA5 async global->LDS (ASYNCcnt) with safe fallback ----------------
#if defined(__AMDGCN__) && __has_builtin(__builtin_amdgcn_global_load_async_to_lds_b128)
#define HAVE_ASYNC 1
#else
#define HAVE_ASYNC 0
#endif
#if defined(__AMDGCN__) && __has_builtin(__builtin_amdgcn_s_wait_asynccnt)
#define HAVE_ASYNC_WAIT_BUILTIN 1
#else
#define HAVE_ASYNC_WAIT_BUILTIN 0
#endif

#if HAVE_ASYNC
typedef __attribute__((address_space(1))) v4i* gp128_t;   // global int4*
typedef __attribute__((address_space(3))) v4i* lp128_t;   // LDS int4*
#endif

__device__ __forceinline__ void lds_copy_b128(u16* dst, const u16* src) {
#if HAVE_ASYNC
    gp128_t g = (gp128_t)(unsigned long long)(uintptr_t)src;
    lp128_t l = (lp128_t)(unsigned)(uintptr_t)dst;
    __builtin_amdgcn_global_load_async_to_lds_b128(g, l, 0, 0);
#else
    *reinterpret_cast<uint4*>(dst) = *reinterpret_cast<const uint4*>(src);
#endif
}
__device__ __forceinline__ void async_wait() {
#if HAVE_ASYNC
#if HAVE_ASYNC_WAIT_BUILTIN
    __builtin_amdgcn_s_wait_asynccnt(0);
#else
    asm volatile("s_wait_asynccnt 0" ::: "memory");
#endif
#endif
}

// native f32 -> bf16 (v_cvt, round-to-nearest-even)
__device__ __forceinline__ u16 f2bf(float f) {
    __bf16 h = (__bf16)f;
    u16 r; __builtin_memcpy(&r, &h, 2);
    return r;
}

__device__ __forceinline__ float gelu_tanh(float u) {
    const float c = 0.7978845608028654f;           // sqrt(2/pi)
    float t = tanhf(c * (u + 0.044715f * u * u * u));
    return 0.5f * u * (1.0f + t);
}

// ---- WMMA fragment loaders: contiguous per-lane runs -> 2x ds_load_b128 ---
// A (16x32 bf16, row-major, row stride `stride`):
//   lane&15 = row M; lanes 0-15: K 0-15; lanes 16-31: K 16-31.
__device__ __forceinline__ v16bf load_a_frag(const u16* __restrict__ base, int stride) {
    int lane = threadIdx.x & 31;
    int row  = lane & 15;
    int kb   = (lane < 16) ? 0 : 16;
    const u16* p = base + (size_t)row * stride;
    BF16Frag f;
#pragma unroll
    for (int e = 0; e < 8; ++e) f.u[e]     = p[kb + e];
#pragma unroll
    for (int e = 0; e < 8; ++e) f.u[8 + e] = p[kb + 8 + e];
    return f.v;
}

// B (32x16 bf16) from an N-major source: logical B[k][n] = base[n*stride + k].
//   lane&15 = col N; lanes 0-15 hold K 0-15, lanes 16-31 hold K 16-31.
__device__ __forceinline__ v16bf load_bT_frag(const u16* __restrict__ base, int stride) {
    int lane = threadIdx.x & 31;
    int n    = lane & 15;
    int kb   = (lane < 16) ? 0 : 16;
    const u16* p = base + (size_t)n * stride + kb;
    BF16Frag f;
#pragma unroll
    for (int e = 0; e < 16; ++e) f.u[e] = p[e];
    return f.v;
}

// ---------------------------------------------------------------------------
// fp32 -> bf16 (plain, for activations / A-side)
// ---------------------------------------------------------------------------
__global__ __launch_bounds__(256) void cvt_f32_bf16(const float* __restrict__ in,
                                                    u16* __restrict__ out, int n) {
    int base = blockIdx.x * 1024 + threadIdx.x;
#pragma unroll
    for (int i = 0; i < 4; ++i) {
        int idx = base + i * 256;
        if (idx < n) out[idx] = f2bf(in[idx]);
    }
}

// fp32 [R][C] -> bf16 transposed [C][R]  (weights -> N-major for GEMM B side)
__global__ __launch_bounds__(256) void cvt_t_f32_bf16(const float* __restrict__ in,
                                                      u16* __restrict__ out,
                                                      int R, int C) {
    __shared__ u16 t[32][33];
    const int bx = blockIdx.x * 32;       // col base (C)
    const int by = blockIdx.y * 32;       // row base (R)
    const int tx = threadIdx.x & 31;
    const int ty = threadIdx.x >> 5;
#pragma unroll
    for (int i = 0; i < 4; ++i) {
        int r = by + ty + i * 8;
        t[ty + i * 8][tx] = f2bf(in[(size_t)r * C + bx + tx]);
    }
    __syncthreads();
#pragma unroll
    for (int i = 0; i < 4; ++i) {
        int c = bx + ty + i * 8;
        out[(size_t)c * R + by + tx] = t[tx][ty + i * 8];
    }
}

// ---------------------------------------------------------------------------
// bf16 GEMM: C[M,N] = (A[M,K] @ B[K,N] + bias) * oscale, B TRANSPOSED (Bt[N][K]).
//   MODE 0: fp32 out    1: bf16 out    2: gelu->bf16 out
//   MODE 3: bf16 out scattered to [b, h, dh, s] (V for attention)
//   Block tile 128x128, BK=32, 4 waves (2x2), 64x64 per wave, double-buffered.
// ---------------------------------------------------------------------------
#define GBM 128
#define GBN 128
#define GBK 32
#define LDT_S 40    // 32 + 8 pad: 80B rows, 16B-aligned b128 accesses

template <int MODE>
__global__ __launch_bounds__(128) void gemm_bf16(const u16* __restrict__ A,
                                                 const u16* __restrict__ Bt,
                                                 const float* __restrict__ bias,
                                                 float* __restrict__ Cf,
                                                 u16* __restrict__ Cb,
                                                 int M, int N, int K,
                                                 float oscale) {
    __shared__ u16 As[2][GBM * LDT_S];
    __shared__ u16 Bs[2][GBN * LDT_S];

    const int tid  = threadIdx.x;
    const int lane = tid & 31;
    const int wave = tid >> 5;            // 4 waves
    const int wm = wave >> 1;             // 0..1 -> 64 rows
    const int wn = wave & 1;              // 0..1 -> 64 cols
    const int brow0 = blockIdx.y * GBM;
    const int bcol0 = blockIdx.x * GBN;

    v8f acc[4][4];
#pragma unroll
    for (int i = 0; i < 4; ++i)
#pragma unroll
        for (int j = 0; j < 4; ++j) acc[i][j] = (v8f){0,0,0,0,0,0,0,0};

    auto stage = [&](int k0, int buf) {
#pragma unroll
        for (int i = 0; i < 4; ++i) {
            int c  = tid + i * 128;       // 0..511
            int r  = c >> 2;
            int kc = (c & 3) * 8;
            lds_copy_b128(&As[buf][r * LDT_S + kc],
                          A  + (size_t)(brow0 + r) * K + k0 + kc);
            lds_copy_b128(&Bs[buf][r * LDT_S + kc],
                          Bt + (size_t)(bcol0 + r) * K + k0 + kc);
        }
    };

    stage(0, 0);
    async_wait();
    __syncthreads();

    int cur = 0;
#pragma unroll 1
    for (int k0 = 0; k0 < K; k0 += GBK) {
        int nxt = cur ^ 1;
        if (k0 + GBK < K) stage(k0 + GBK, nxt);

        v16bf bfr[4];
#pragma unroll
        for (int j = 0; j < 4; ++j)
            bfr[j] = load_bT_frag(&Bs[cur][(wn * 64 + j * 16) * LDT_S], LDT_S);
#pragma unroll
        for (int i = 0; i < 4; ++i) {
            v16bf afr = load_a_frag(&As[cur][(wm * 64 + i * 16) * LDT_S], LDT_S);
#pragma unroll
            for (int j = 0; j < 4; ++j)
                acc[i][j] = __builtin_amdgcn_wmma_f32_16x16x32_bf16(
                    false, afr, false, bfr[j], (short)0, acc[i][j], false, false);
        }
        async_wait();
        __syncthreads();
        cur = nxt;
    }

    // --- epilogue (branch-free per instantiation) --------------------------
    const int hi8 = (lane < 16) ? 0 : 8;
#pragma unroll
    for (int i = 0; i < 4; ++i) {
#pragma unroll
        for (int j = 0; j < 4; ++j) {
            int col = bcol0 + wn * 64 + j * 16 + (lane & 15);
            float bv = bias[col];
            int rbase = brow0 + wm * 64 + i * 16 + hi8;
#pragma unroll
            for (int r = 0; r < 8; ++r) {
                float val = (acc[i][j][r] + bv) * oscale;
                int row = rbase + r;
                if (MODE == 0) {
                    Cf[(size_t)row * N + col] = val;
                } else if (MODE == 1) {
                    Cb[(size_t)row * N + col] = f2bf(val);
                } else if (MODE == 2) {
                    Cb[(size_t)row * N + col] = f2bf(gelu_tanh(val));
                } else {              // MODE 3: [b,h,dh,s] layout for V
                    int b = row >> 10, s = row & 1023;
                    int h = col >> 6,  dh = col & 63;
                    size_t idx = ((size_t)((b * NH + h) * DH + dh) << 10) | s;
                    Cb[idx] = f2bf(val);
                }
            }
        }
    }
}

// ---------------------------------------------------------------------------
// Flash attention: one block (8 waves) per (b, h, 128-query-row slab).
// K chunk staged row-major [key][dh]; V staged from transposed [dh][key].
// Double-buffered staging; P packed to bf16 A-layout in LDS by softmax.
// 1/sqrt(dh) is already folded into Q.
// ---------------------------------------------------------------------------
#define KV_S 72     // 64 + 8 pad (144B rows)

__global__ __launch_bounds__(256) void attn_flash(const u16* __restrict__ q,
                                                  const u16* __restrict__ k,
                                                  const u16* __restrict__ vT,
                                                  u16* __restrict__ o) {
    __shared__ u16   Kt[2][32 * KV_S];
    __shared__ u16   Vt[2][64 * LDT_S];      // [dh][key] chunk, stride 40
    __shared__ float Sst[8][16][33];         // per-wave scores (f32)
    __shared__ u16   Pst[8][16 * LDT_S];     // per-wave P in bf16 A-layout
    __shared__ float astat[8][16];
    __shared__ float lstat[8][16];

    const int bid  = blockIdx.x;
    const int qblk = bid & 7;
    const int h    = (bid >> 3) & (NH - 1);
    const int b    = bid >> 7;
    const int tid  = threadIdx.x;
    const int w    = tid >> 5;
    const int lane = tid & 31;
    const int hi8  = (lane < 16) ? 0 : 8;
    const int q0   = qblk * 128 + w * 16;

    const u16* qbase = q + ((size_t)b * SEQ + q0) * DIM + h * DH;
    v16bf aQ0 = load_a_frag(qbase, DIM);
    v16bf aQ1 = load_a_frag(qbase + 32, DIM);

    const size_t kbase  = ((size_t)b * SEQ) * DIM + h * DH;
    const size_t vtbase = ((size_t)(b * NH + h) * DH) << 10;   // [dh][s]

    auto stageKV = [&](int c0, int buf) {
        {   // K: 32 keys x 64 dh, 256 b128 chunks
            int r  = tid >> 3;
            int cc = (tid & 7) * 8;
            lds_copy_b128(&Kt[buf][r * KV_S + cc],
                          k + kbase + (size_t)(c0 + r) * DIM + cc);
        }
        {   // V^T: 64 dh x 32 keys, 256 b128 chunks
            int r  = tid >> 2;
            int kc = (tid & 3) * 8;
            lds_copy_b128(&Vt[buf][r * LDT_S + kc],
                          vT + vtbase + ((size_t)r << 10) + c0 + kc);
        }
    };

    v8f O[4];
#pragma unroll
    for (int t = 0; t < 4; ++t) O[t] = (v8f){0,0,0,0,0,0,0,0};
    float mrow = -1e30f, lrow = 0.f;         // lanes 0..15 own one row each

    stageKV(0, 0);
    async_wait();
    __syncthreads();

    int cur = 0;
#pragma unroll 1
    for (int c0 = 0; c0 < SEQ; c0 += 32) {
        int nxt = cur ^ 1;
        if (c0 + 32 < SEQ) stageKV(c0 + 32, nxt);

        // ---- scores: two 16x16 tiles, dh=64 contraction (2 WMMAs each)
#pragma unroll
        for (int t = 0; t < 2; ++t) {
            v16bf bk0 = load_bT_frag(&Kt[cur][(t * 16) * KV_S], KV_S);
            v16bf bk1 = load_bT_frag(&Kt[cur][(t * 16) * KV_S + 32], KV_S);
            v8f z = (v8f){0,0,0,0,0,0,0,0};
            z = __builtin_amdgcn_wmma_f32_16x16x32_bf16(false, aQ0, false, bk0,
                                                        (short)0, z, false, false);
            z = __builtin_amdgcn_wmma_f32_16x16x32_bf16(false, aQ1, false, bk1,
                                                        (short)0, z, false, false);
#pragma unroll
            for (int r = 0; r < 8; ++r)
                Sst[w][r + hi8][t * 16 + (lane & 15)] = z[r];
        }
        __syncthreads();

        // ---- online softmax: lanes 0..15, P emitted pre-packed bf16
        if (lane < 16) {
            float mc = -1e30f;
#pragma unroll
            for (int c = 0; c < 32; ++c) mc = fmaxf(mc, Sst[w][lane][c]);
            float nm = fmaxf(mrow, mc);
            float alpha = __expf(mrow - nm);
            float sum = 0.f;
#pragma unroll
            for (int c = 0; c < 32; c += 2) {
                float p0 = __expf(Sst[w][lane][c]     - nm);
                float p1 = __expf(Sst[w][lane][c + 1] - nm);
                sum += p0 + p1;
                unsigned pk = (unsigned)f2bf(p0) | ((unsigned)f2bf(p1) << 16);
                *reinterpret_cast<unsigned*>(&Pst[w][lane * LDT_S + c]) = pk;
            }
            lrow = lrow * alpha + sum;
            mrow = nm;
            astat[w][lane] = alpha;
        }
        __syncthreads();

        float ar[8];
#pragma unroll
        for (int r = 0; r < 8; ++r) ar[r] = astat[w][r + hi8];
        v16bf pf = load_a_frag(&Pst[w][0], LDT_S);

        // ---- O = diag(alpha) * O + P @ V
#pragma unroll
        for (int t2 = 0; t2 < 4; ++t2) {
            v16bf vf = load_bT_frag(&Vt[cur][(t2 * 16) * LDT_S], LDT_S);
            v8f on = O[t2];
#pragma unroll
            for (int r = 0; r < 8; ++r) on[r] *= ar[r];
            O[t2] = __builtin_amdgcn_wmma_f32_16x16x32_bf16(
                false, pf, false, vf, (short)0, on, false, false);
        }
        async_wait();
        __syncthreads();
        cur = nxt;
    }

    if (lane < 16) lstat[w][lane] = lrow;
    __syncthreads();

    float linv[8];
#pragma unroll
    for (int r = 0; r < 8; ++r) linv[r] = 1.f / lstat[w][r + hi8];
#pragma unroll
    for (int t2 = 0; t2 < 4; ++t2) {
#pragma unroll
        for (int r = 0; r < 8; ++r) {
            size_t idx = ((size_t)b * SEQ + q0 + r + hi8) * DIM
                       + h * DH + t2 * 16 + (lane & 15);
            o[idx] = f2bf(O[t2][r] * linv[r]);
        }
    }
}

// ---------------------------------------------------------------------------
// u = x + y ; LayerNorm(u) * g + b  -> fp32 out (and optional bf16 copy)
// ---------------------------------------------------------------------------
__global__ __launch_bounds__(256) void add_ln(const float* __restrict__ x,
                                              const float* __restrict__ y,
                                              const float* __restrict__ g,
                                              const float* __restrict__ bb,
                                              float* __restrict__ outf,
                                              u16* __restrict__ outb) {
    __shared__ float red[256];
    const int row = blockIdx.x;
    const int tid = threadIdx.x;
    const float* xr = x + (size_t)row * DIM;
    const float* yr = y + (size_t)row * DIM;

    float u[4];
    float s = 0.f;
#pragma unroll
    for (int i = 0; i < 4; ++i) {
        u[i] = xr[tid + i * 256] + yr[tid + i * 256];
        s += u[i];
    }
    red[tid] = s; __syncthreads();
    for (int off = 128; off > 0; off >>= 1) {
        if (tid < off) red[tid] += red[tid + off];
        __syncthreads();
    }
    float mean = red[0] * (1.f / DIM);
    __syncthreads();

    float s2 = 0.f;
#pragma unroll
    for (int i = 0; i < 4; ++i) { float d = u[i] - mean; s2 += d * d; }
    red[tid] = s2; __syncthreads();
    for (int off = 128; off > 0; off >>= 1) {
        if (tid < off) red[tid] += red[tid + off];
        __syncthreads();
    }
    float rstd = rsqrtf(red[0] * (1.f / DIM) + 1e-5f);

#pragma unroll
    for (int i = 0; i < 4; ++i) {
        int c = tid + i * 256;
        float o = (u[i] - mean) * rstd * g[c] + bb[c];
        size_t idx = (size_t)row * DIM + c;
        if (outf) outf[idx] = o;
        if (outb) outb[idx] = f2bf(o);
    }
}

// ---------------------------------------------------------------------------
// host-side orchestration
// ---------------------------------------------------------------------------
extern "C" void kernel_launch(void* const* d_in, const int* in_sizes, int n_in,
                              void* d_out, int out_size, void* d_ws, size_t ws_size,
                              hipStream_t stream) {
    const float* x   = (const float*)d_in[0];
    const float* Wq  = (const float*)d_in[1];
    const float* bq  = (const float*)d_in[2];
    const float* Wk  = (const float*)d_in[3];
    const float* bk  = (const float*)d_in[4];
    const float* Wv  = (const float*)d_in[5];
    const float* bv  = (const float*)d_in[6];
    const float* Wo  = (const float*)d_in[7];
    const float* bo  = (const float*)d_in[8];
    const float* g1  = (const float*)d_in[9];
    const float* b1  = (const float*)d_in[10];
    const float* W1  = (const float*)d_in[11];
    const float* bm1 = (const float*)d_in[12];
    const float* W2  = (const float*)d_in[13];
    const float* bm2 = (const float*)d_in[14];
    const float* g2  = (const float*)d_in[15];
    const float* b2  = (const float*)d_in[16];
    float* out = (float*)d_out;

    char* ws = (char*)d_ws;
    const size_t MBY = 1024 * 1024;
    // workspace layout (104 MB total; hb aliases dead xb/qb/kb/vbT region)
    u16*   WqT  = (u16*)(ws + 0 * MBY);    // transposed bf16 weights [N][K]
    u16*   WkT  = (u16*)(ws + 2 * MBY);
    u16*   WvT  = (u16*)(ws + 4 * MBY);
    u16*   WoT  = (u16*)(ws + 6 * MBY);
    u16*   W1T  = (u16*)(ws + 8 * MBY);
    u16*   W2T  = (u16*)(ws + 16 * MBY);
    u16*   xb   = (u16*)(ws + 24 * MBY);
    u16*   qb   = (u16*)(ws + 32 * MBY);
    u16*   kb   = (u16*)(ws + 40 * MBY);
    u16*   vbT  = (u16*)(ws + 48 * MBY);   // [b,h,dh,s]
    u16*   ab   = (u16*)(ws + 56 * MBY);
    u16*   x1b  = (u16*)(ws + 64 * MBY);
    float* x1f  = (float*)(ws + 72 * MBY);
    float* fraw = (float*)(ws + 88 * MBY);
    u16*   hb   = (u16*)(ws + 24 * MBY);   // 32 MB, reuses xb..vbT after attention

    auto cvt = [&](const float* src, u16* dst, int n) {
        cvt_f32_bf16<<<(n + 1023) / 1024, 256, 0, stream>>>(src, dst, n);
    };
    auto cvtT = [&](const float* src, u16* dst, int R, int C) {
        dim3 grid(C / 32, R / 32);
        cvt_t_f32_bf16<<<grid, 256, 0, stream>>>(src, dst, R, C);
    };

    // 1. activations plain-converted; weights transpose-converted (N-major)
    cvt (x,  xb,  TOK * DIM);
    cvtT(Wq, WqT, DIM, DIM);
    cvtT(Wk, WkT, DIM, DIM);
    cvtT(Wv, WvT, DIM, DIM);
    cvtT(Wo, WoT, DIM, DIM);
    cvtT(W1, W1T, DIM, NDFF);
    cvtT(W2, W2T, NDFF, DIM);

    // 2. QKV projections (Q pre-scaled by 1/sqrt(DH))
    {
        dim3 gd(DIM / GBN, TOK / GBM);
        gemm_bf16<1><<<gd, 128, 0, stream>>>(xb, WqT, bq, nullptr, qb,  TOK, DIM, DIM, 0.125f);
        gemm_bf16<1><<<gd, 128, 0, stream>>>(xb, WkT, bk, nullptr, kb,  TOK, DIM, DIM, 1.0f);
        gemm_bf16<3><<<gd, 128, 0, stream>>>(xb, WvT, bv, nullptr, vbT, TOK, DIM, DIM, 1.0f);
    }

    // 3. flash attention: 4 batches * 16 heads * 8 slabs = 512 blocks
    attn_flash<<<NB * NH * 8, 256, 0, stream>>>(qb, kb, vbT, ab);

    // 4. output projection -> fp32 raw
    {
        dim3 gd(DIM / GBN, TOK / GBM);
        gemm_bf16<0><<<gd, 128, 0, stream>>>(ab, WoT, bo, fraw, nullptr, TOK, DIM, DIM, 1.0f);
    }

    // 5. x1 = LN(x + attn) -> fp32 (residual) + bf16 (GEMM input)
    add_ln<<<TOK, 256, 0, stream>>>(x, fraw, g1, b1, x1f, x1b);

    // 6. MLP up: gelu(x1 @ W1 + bm1) -> bf16
    {
        dim3 gd(NDFF / GBN, TOK / GBM);
        gemm_bf16<2><<<gd, 128, 0, stream>>>(x1b, W1T, bm1, nullptr, hb, TOK, NDFF, DIM, 1.0f);
    }

    // 7. MLP down: h @ W2 + bm2 -> fp32 raw
    {
        dim3 gd(DIM / GBN, TOK / GBM);
        gemm_bf16<0><<<gd, 128, 0, stream>>>(hb, W2T, bm2, fraw, nullptr, TOK, DIM, NDFF, 1.0f);
    }

    // 8. out = LN(x1 + mlp)
    add_ln<<<TOK, 256, 0, stream>>>(x1f, fraw, g2, b2, out, nullptr);
}
